// DispInit_30872224924310
// MI455X (gfx1250) — matrix operand order repaired and checked by reading, not compiled, not run on hardware
//
#include <hip/hip_runtime.h>
#include <hip/hip_bf16.h>

typedef __attribute__((ext_vector_type(16))) _Float16 v16h;
typedef __attribute__((ext_vector_type(8)))  _Float16 v8h;
typedef __attribute__((ext_vector_type(8)))  float    v8f;
typedef __attribute__((ext_vector_type(4)))  unsigned int u32x4;
typedef __attribute__((ext_vector_type(8)))  int      i32x8;
typedef __attribute__((ext_vector_type(4)))  int      i32x4;

#ifndef __has_builtin
#define __has_builtin(x) 0
#endif

#if __has_builtin(__builtin_amdgcn_tensor_load_to_lds)
#define HAVE_TDM 1
#else
#define HAVE_TDM 0
#endif

// ---------------------------------------------------------------------------
// Kernel 1: LayerNorm over channels (C=128), transpose (b,c,h,i)->(b,h,i,c),
// emit FP16 for the WMMA GEMMs. One block = (b, h, 32-wide i-tile).
// ---------------------------------------------------------------------------
__global__ __launch_bounds__(128)
void ln_f16_kernel(const float* __restrict__ F,
                   const float* __restrict__ lnw,
                   const float* __restrict__ lnb,
                   _Float16* __restrict__ X)
{
    int blk = blockIdx.x;
    int it  = blk & 7;        // i-tile (256/32)
    int bh  = blk >> 3;       // b*80 + h
    int b   = bh / 80;
    int h   = bh % 80;
    int c   = threadIdx.x;    // 0..127
    int i0  = it * 32;

    __shared__ float xs[32][132];
    __shared__ float ps[4][32], pq[4][32];
    __shared__ float smu[32], srs[32];

    const float* src = F + (((size_t)(b * 128 + c) * 80 + h) * 256 + i0);
    float v[32];
#pragma unroll
    for (int k = 0; k < 8; ++k) {
        float4 q = *(const float4*)(src + 4 * k);
        v[4*k+0] = q.x; v[4*k+1] = q.y; v[4*k+2] = q.z; v[4*k+3] = q.w;
    }
#pragma unroll
    for (int il = 0; il < 32; ++il) xs[il][c] = v[il];
    __syncthreads();

    int il = threadIdx.x & 31;
    int cq = threadIdx.x >> 5;
    float s = 0.f, sq = 0.f;
#pragma unroll
    for (int cc = 0; cc < 32; ++cc) {
        float x = xs[il][cq * 32 + cc];
        s += x; sq += x * x;
    }
    ps[cq][il] = s; pq[cq][il] = sq;
    __syncthreads();

    if (threadIdx.x < 32) {
        int i = threadIdx.x;
        float sum = ps[0][i] + ps[1][i] + ps[2][i] + ps[3][i];
        float ssq = pq[0][i] + pq[1][i] + pq[2][i] + pq[3][i];
        float mu  = sum * (1.0f / 128.0f);
        float var = ssq * (1.0f / 128.0f) - mu * mu;
        smu[i] = mu;
        srs[i] = rsqrtf(var + 1e-5f);
    }
    __syncthreads();

    float wc = lnw[c], bc = lnb[c];
    _Float16* dst = X + (((size_t)bh * 256 + i0) * 128 + c);
#pragma unroll
    for (int k = 0; k < 32; ++k) {
        float y = (xs[k][c] - smu[k]) * srs[k] * wc + bc;
        dst[(size_t)k * 128] = (_Float16)y;
    }
}

// ---------------------------------------------------------------------------
// TDM descriptor packing (CDNA5 ISA 8.3-8.6). 2D tile of `tileRows` rows x
// 128 fp16 elements, row stride `rowStride` elements, loaded into LDS with
// a 16B pad after every 256B row (pad_interval=5 -> 64 DW, pad_amount=3 ->
// 4 DW) giving a 136-half (272B) LDS row stride for bank-conflict-free
// fragment reads.
// ---------------------------------------------------------------------------
__device__ __forceinline__ u32x4 tdm_group0(unsigned lds_off, unsigned long long gaddr)
{
    u32x4 g;
    g[0] = 1u;                                              // count=1 (valid user D#)
    g[1] = lds_off;                                         // lds_addr [63:32]
    g[2] = (unsigned)(gaddr & 0xffffffffull);               // global_addr lo
    g[3] = (unsigned)((gaddr >> 32) & 0x01ffffffull) | (2u << 30); // addr hi | type=2
    return g;
}

__device__ __forceinline__ i32x8 tdm_group1(unsigned tileRows, unsigned rowStride)
{
    // data_size=1 (2B), pad_enable, pad_interval=5 (64 DW=256B), pad_amount=3 (4 DW=16B)
    unsigned w0   = (1u << 16) | (1u << 20) | (5u << 22) | (3u << 25);
    unsigned d0   = 128u;            // tensor_dim0 (elements per row)
    unsigned d1   = tileRows;        // tensor_dim1 (rows)
    unsigned t0   = 128u;            // tile_dim0
    i32x8 g;
    g[0] = (int)w0;
    g[1] = (int)((d0 & 0xffffu) << 16);                     // abar_addr=0 | dim0.lo16
    g[2] = (int)((d0 >> 16) | ((d1 & 0xffffu) << 16));      // dim0.hi16 | dim1.lo16
    g[3] = (int)((d1 >> 16) | (t0 << 16));                  // dim1.hi16 | tile_dim0
    g[4] = (int)(tileRows & 0xffffu);                       // tile_dim1 | tile_dim2=0
    g[5] = (int)rowStride;                                  // tensor_dim0_stride lo32
    g[6] = 0;                                               // stride hi16 | dim1_stride lo
    g[7] = 0;
    return g;
}

// ---------------------------------------------------------------------------
// Kernel 2: correlation GEMM OUT[slice,m,n] = sum_k F0[row_m,k]*F1[row_n,k]
// (A·Bt, K=128) via v_wmma_f32_16x16x32_f16.
//   Block: 256 threads (8 waves), macro-tile 128(M) x 64(N) staged in LDS
//   via the Tensor Data Mover (fallback: vectorized copy). Each wave owns a
//   32x32 register tile => 4 accumulators, 16 WMMAs per wave.
// Fragment layouts per ISA 7.12.2 (wave32):
//   A: lanes 0-15 -> M=lane, K={0..7,16..23}; lanes 16-31 -> K={8..15,24..31}
//   B: lanes 0-15 -> N=lane, K={0..7,16..23};         lanes 16-31 -> K={8..15,24..31}
//   C: VGPR r -> M=(lane>>4)*8+r, N=lane&15
// ---------------------------------------------------------------------------
#define LDS_ROW   136                       // 128 halves + 16B pad (272B stride)
#define A_HALVES  (128 * LDS_ROW)           // 34816B
#define B_HALVES  (64  * LDS_ROW)

template <int HOUT, int MOUT, int HSTEP, int RSTEP>
__global__ __launch_bounds__(256)
void cv_wmma_tiled(const _Float16* __restrict__ F0,
                   const _Float16* __restrict__ F1,
                   float* __restrict__ OUT)
{
    constexpr int NT = MOUT / 64;           // macro-tiles along N
    constexpr int MT = MOUT / 128;          // macro-tiles along M

    __shared__ _Float16 smem[A_HALVES + B_HALVES];
    _Float16* As = smem;
    _Float16* Bs = smem + A_HALVES;

    int blk   = blockIdx.x;
    int slice = blk / (MT * NT);
    int rem   = blk % (MT * NT);
    int tM    = rem / NT;
    int tN    = rem % NT;
    int b     = slice / HOUT;
    int h     = slice % HOUT;

    size_t srcSlice = ((size_t)b * 80 + (size_t)h * HSTEP) * 256;
    const _Float16* gA = F0 + (srcSlice + (size_t)(tM * 128) * RSTEP) * 128;
    const _Float16* gB = F1 + (srcSlice + (size_t)(tN * 64)  * RSTEP) * 128;

#if HAVE_TDM
    if (threadIdx.x < 32) {                 // wave-uniform: one wave drives the TDM
        u32x4 a0 = tdm_group0(0u,                      (unsigned long long)gA);
        i32x8 a1 = tdm_group1(128u, (unsigned)(RSTEP * 128));
        u32x4 b0 = tdm_group0((unsigned)(A_HALVES * 2), (unsigned long long)gB);
        i32x8 b1 = tdm_group1(64u,  (unsigned)(RSTEP * 128));
        i32x4 z4 = {0, 0, 0, 0};
#if defined(__clang_major__) && (__clang_major__ >= 23)
        i32x8 z8 = {0, 0, 0, 0, 0, 0, 0, 0};
        __builtin_amdgcn_tensor_load_to_lds(a0, a1, z4, z4, z8, 0);
        __builtin_amdgcn_tensor_load_to_lds(b0, b1, z4, z4, z8, 0);
#else
        __builtin_amdgcn_tensor_load_to_lds(a0, a1, z4, z4, 0);
        __builtin_amdgcn_tensor_load_to_lds(b0, b1, z4, z4, 0);
#endif
#if __has_builtin(__builtin_amdgcn_s_wait_tensorcnt)
        __builtin_amdgcn_s_wait_tensorcnt((short)0);
#else
        asm volatile("s_wait_tensorcnt 0x0" ::: "memory");
#endif
    }
    __syncthreads();
#else
    // Fallback: vectorized staged copy into the same padded LDS layout.
    for (int c = threadIdx.x; c < 128 * 16; c += 256) {
        int r = c >> 4, q = c & 15;
        float4 t = *(const float4*)(gA + (size_t)r * RSTEP * 128 + q * 8);
        *(float4*)(As + (size_t)r * LDS_ROW + q * 8) = t;
    }
    for (int c = threadIdx.x; c < 64 * 16; c += 256) {
        int r = c >> 4, q = c & 15;
        float4 t = *(const float4*)(gB + (size_t)r * RSTEP * 128 + q * 8);
        *(float4*)(Bs + (size_t)r * LDS_ROW + q * 8) = t;
    }
    __syncthreads();
#endif

    int lane = threadIdx.x & 31;
    int wave = threadIdx.x >> 5;
    int wm   = wave >> 1;                   // 0..3 : 32-row group
    int wn   = wave & 1;                    // 0..1 : 32-col group
    int l15  = lane & 15;
    int lh   = lane >> 4;                   // 0 or 1

    v8f acc[2][2] = {};
#pragma unroll
    for (int k0 = 0; k0 < 128; k0 += 32) {
        v16h af[2], bf[2];
#pragma unroll
        for (int t2 = 0; t2 < 2; ++t2) {
            const _Float16* ar = As + (size_t)(wm * 32 + t2 * 16 + l15) * LDS_ROW
                                    + k0 + lh * 8;
            v8h alo = *(const v8h*)ar;              // K = k0+lh*8 .. +7
            v8h ahi = *(const v8h*)(ar + 16);       // K = k0+lh*8+16 .. +23
#pragma unroll
            for (int q = 0; q < 8; ++q) { af[t2][q] = alo[q]; af[t2][8 + q] = ahi[q]; }

            const _Float16* br = Bs + (size_t)(wn * 32 + t2 * 16 + l15) * LDS_ROW
                                    + k0 + lh * 8;
            v8h blo = *(const v8h*)br;              // K = k0+lh*8 .. +7
            v8h bhi = *(const v8h*)(br + 16);       // K = k0+lh*8+16 .. +23
#pragma unroll
            for (int q = 0; q < 8; ++q) { bf[t2][q] = blo[q]; bf[t2][8 + q] = bhi[q]; }
        }
#pragma unroll
        for (int m2 = 0; m2 < 2; ++m2)
#pragma unroll
            for (int n2 = 0; n2 < 2; ++n2)
                acc[m2][n2] = __builtin_amdgcn_wmma_f32_16x16x32_f16(
                    false, af[m2], false, bf[n2],
                    (short)0, acc[m2][n2], false, false);
    }

    float* obase = OUT + (size_t)slice * MOUT * MOUT;
#pragma unroll
    for (int m2 = 0; m2 < 2; ++m2) {
#pragma unroll
        for (int n2 = 0; n2 < 2; ++n2) {
            int gm = tM * 128 + wm * 32 + m2 * 16 + lh * 8;
            int gn = tN * 64  + wn * 32 + n2 * 16 + l15;
            float* o = obase + (size_t)gm * MOUT + gn;
#pragma unroll
            for (int r = 0; r < 8; ++r) o[(size_t)r * MOUT] = acc[m2][n2][r];
        }
    }
}

// ---------------------------------------------------------------------------
// Kernel 3: fused masked Sinkhorn (8 iters, 257x257 log-domain) + finalize.
// One block per (b,h) slice; u,v in LDS; online (rescaling) logsumexp so the
// -inf masked entries are simply skipped.
// ---------------------------------------------------------------------------
__device__ __forceinline__ void lse_acc(float& m, float& s, float x)
{
    if (x > m) { s = s * __expf(m - x) + 1.0f; m = x; }
    else       { s += __expf(x - m); }
}

__global__ __launch_bounds__(256)
void sinkhorn_kernel(const float* __restrict__ CV,
                     float* __restrict__ disp,
                     float* __restrict__ conf,
                     float* __restrict__ occ)
{
    const int   W      = 256;
    const float LOGMU  = -6.23832463f;   // -log(2w)
    const float LOGMUE = -0.69314718f;   // log(0.5)
    const float LOG2W  =  6.23832463f;   // log(2w)

    int slice = blockIdx.x;              // b*80 + h
    const float* C = CV + (size_t)slice * W * W;
    int t = threadIdx.x;

    __shared__ float su[257];
    __shared__ float sv[257];

    for (int i = t; i < 257; i += 256) su[i] = 0.0f;
    __syncthreads();

    for (int it = 0; it < 8; ++it) {
        for (int j = t; j < 257; j += 256) {         // v-pass (columns, coalesced)
            float m = -INFINITY, s = 0.0f;
            if (j < W) {
                for (int i = j; i < W; ++i)
                    lse_acc(m, s, C[(size_t)i * W + j] + su[i]);
                lse_acc(m, s, su[256]);
                sv[j] = LOGMU - (m + __logf(s));
            } else {
                for (int i = 0; i < 257; ++i) lse_acc(m, s, su[i]);
                sv[j] = LOGMUE - (m + __logf(s));
            }
        }
        __syncthreads();
        for (int i = t; i < 257; i += 256) {         // u-pass (rows)
            float m = -INFINITY, s = 0.0f;
            if (i < W) {
                const float* row = C + (size_t)i * W;
                for (int j = 0; j <= i; ++j) lse_acc(m, s, row[j] + sv[j]);
                lse_acc(m, s, sv[256]);
                su[i] = LOGMU - (m + __logf(s));
            } else {
                for (int j = 0; j < 257; ++j) lse_acc(m, s, sv[j]);
                su[i] = LOGMUE - (m + __logf(s));
            }
        }
        __syncthreads();
    }

    for (int i = t; i < W; i += 256) {
        const float* row = C + (size_t)i * W;
        float ui = su[i] + LOG2W;
        float best = -1.0f;
        int   jm   = 0;
        float sum  = 0.0f;
        for (int j = 0; j <= i; ++j) {
            float p = __expf(row[j] + ui + sv[j]);
            sum += p;
            if (p > best) { best = p; jm = j; }
        }
        size_t oidx = (size_t)slice * W + i;
        occ[oidx] = sum;

        float cf = 0.0f, cr = 0.0f;
#pragma unroll
        for (int d = -2; d <= 2; ++d) {
            int jj = jm + d;
            float wgt = (jj >= 0 && jj <= i) ? __expf(row[jj] + ui + sv[jj]) : 0.0f;
            cf += wgt;
            cr += wgt * (float)jj;
        }
        cr = (cr + 1e-4f) / (cf + 1e-4f);
        disp[oidx] = (float)i - cr;
        conf[oidx] = cf;
    }
}

// ---------------------------------------------------------------------------
// Launch: outputs concatenated as (disparity, conf, occ, cv, cv_down).
// ---------------------------------------------------------------------------
extern "C" void kernel_launch(void* const* d_in, const int* in_sizes, int n_in,
                              void* d_out, int out_size, void* d_ws, size_t ws_size,
                              hipStream_t stream)
{
    const float* feature = (const float*)d_in[0];   // (8,128,80,256)
    const float* lnw     = (const float*)d_in[1];   // (128,)
    const float* lnb     = (const float*)d_in[2];   // (128,)

    float* out  = (float*)d_out;
    float* disp = out;                              // 4*1*80*256   = 81920
    float* conf = out + 81920;
    float* occ  = out + 163840;
    float* cv   = out + 245760;                     // 4*80*256*256 = 20971520
    float* cvd  = out + 245760 + 20971520;          // 4*40*128*128 = 2621440

    _Float16* xh  = (_Float16*)d_ws;                // (8,80,256,128) fp16, 42 MB
    _Float16* f0h = xh;
    _Float16* f1h = xh + (size_t)4 * 80 * 256 * 128;

    // 1) LayerNorm + fp16 pack.
    ln_f16_kernel<<<8 * 80 * 8, 128, 0, stream>>>(feature, lnw, lnb, xh);

    // 2) cv: 320 slices * (256/128)*(256/64) = 2560 macro-tile blocks.
    cv_wmma_tiled<80, 256, 1, 1><<<320 * 2 * 4, 256, 0, stream>>>(f0h, f1h, cv);

    // 3) cv_down: 160 slices * (128/128)*(128/64) = 320 blocks.
    cv_wmma_tiled<40, 128, 2, 2><<<160 * 1 * 2, 256, 0, stream>>>(f0h, f1h, cvd);

    // 4) Sinkhorn + finalize: one block per (b,h) slice.
    sinkhorn_kernel<<<4 * 80, 256, 0, stream>>>(cv, disp, conf, occ);
}
